// PaperMultiRelationGATLayer_88751204204687
// MI455X (gfx1250) — compile-verified, matrix operand order checked by reading, not run on hardware
//
#include <hip/hip_runtime.h>

// ---------------------------------------------------------------------------
// PaperMultiRelationGATLayer, factorized: r_ij depends only on the type
// triple (e_ij, u_i, u_j) -> precompute 1000-row combo tables with bf16 WMMA
// GEMMs, then a gather/softmax/aggregate pass against L2-resident tables.
// B=8, N=128, D=256, R=40, T=5.
// ---------------------------------------------------------------------------

constexpr int DD  = 256;           // feature dim
constexpr int BB  = 8;             // batch
constexpr int NN  = 128;           // nodes
constexpr int RR  = 40;            // edge types
constexpr int TT  = 5;             // node types
constexpr int NC  = RR * TT * TT;  // 1000 combos
constexpr int NCP = 1008;          // padded to 16-row tiles
constexpr int MN  = BB * NN;       // 1024 node rows

typedef __attribute__((ext_vector_type(16))) __bf16 v16bf;
typedef __attribute__((ext_vector_type(8)))  float  v8f;

union FragBF { v16bf v; unsigned int u[8]; };

// K offset inside a 32-wide K chunk for VGPR v, lane-half hl (16-bit operands).
__device__ __forceinline__ int frag_koff(int v, int hl) {
  return ((v & 4) << 2) + (hl << 3) + ((v & 3) << 1);
}

// ---------------------------------------------------------------------------
// C[M,256](f32 and/or bf16) = A[M,256](bf16) @ Wt^T + bias (+ residual)
// Wt is [256 out][256 in] bf16 (pre-transposed so B-operand K pairs are one
// dword). One wave = one 16x16 tile, 8 x v_wmma_f32_16x16x32_bf16 over K=256.
// grid = (M/16, 2), block = (32, 8); 16x256 A tile staged in LDS per block.
// ---------------------------------------------------------------------------
__global__ __launch_bounds__(256)
void k_gemm_bf16(const __bf16* __restrict__ A, const __bf16* __restrict__ Wt,
                 const float* __restrict__ bias, const float* __restrict__ residual,
                 float* __restrict__ outF, __bf16* __restrict__ outB) {
  __shared__ __bf16 As[16 * DD];
  const int m0   = blockIdx.x * 16;
  const int lane = threadIdx.x;
  const int wave = threadIdx.y;
  const int n0   = (blockIdx.y * 8 + wave) * 16;

  {  // stage A tile: 16*256 bf16 = 2048 dwords, 8 per thread, coalesced
    const unsigned int* src = (const unsigned int*)(A + (size_t)m0 * DD);
    unsigned int* dst = (unsigned int*)As;
    const int tid = wave * 32 + lane;
#pragma unroll
    for (int i = 0; i < (16 * DD / 2) / 256; ++i)
      dst[tid + i * 256] = src[tid + i * 256];
  }
  __syncthreads();

  const int rc = lane & 15;   // row of A-frag / col of B-frag and C tile
  const int hl = lane >> 4;

  v8f acc = {};
#pragma unroll
  for (int kk = 0; kk < DD / 32; ++kk) {
    FragBF a, b;
#pragma unroll
    for (int v = 0; v < 8; ++v) {
      const int k = kk * 32 + frag_koff(v, hl);
      a.u[v] = *(const unsigned int*)&As[rc * DD + k];
      b.u[v] = *(const unsigned int*)&Wt[(size_t)(n0 + rc) * DD + k];
    }
    acc = __builtin_amdgcn_wmma_f32_16x16x32_bf16(
        /*neg_a=*/false, a.v, /*neg_b=*/false, b.v,
        /*c_mod=*/(short)0, acc, /*reuse_a=*/false, /*reuse_b=*/false);
  }

#pragma unroll
  for (int g = 0; g < 8; ++g) {
    const int m = m0 + hl * 8 + g;   // C/D layout: VGPR g -> row g (+8 upper half)
    const int n = n0 + rc;
    float val = acc[g];
    if (bias)     val += bias[n];
    if (residual) val += residual[(size_t)m * DD + n];
    if (outF) outF[(size_t)m * DD + n] = val;
    if (outB) outB[(size_t)m * DD + n] = (__bf16)val;
  }
}

// Transpose+convert a (256-in x 256-out) f32 weight block (rows koff..koff+255)
// into bf16 [out][in].
__global__ void k_transpose_bf16(const float* __restrict__ W, int koff,
                                 __bf16* __restrict__ Wt) {
  const int n = blockIdx.x, k = threadIdx.x;
  Wt[(size_t)n * DD + k] = (__bf16)W[(size_t)(koff + k) * DD + n];
}

__global__ void k_cvt_bf16(const float* __restrict__ in, __bf16* __restrict__ out, int n) {
  const int i = blockIdx.x * blockDim.x + threadIdx.x;
  if (i < n) out[i] = (__bf16)in[i];
}

// h1[c] = relu(W1[e] + W1[R+us] + W1[R+T+ud] + b1) for combo c=(e*T+us)*T+ud.
__global__ void k_build_h1(const float* __restrict__ W1, const float* __restrict__ b1,
                           __bf16* __restrict__ h1) {
  const int c = blockIdx.x, t = threadIdx.x;
  float v = 0.f;
  if (c < NC) {
    const int e  = c / (TT * TT);
    const int us = (c / TT) % TT;
    const int ud = c % TT;
    v = W1[e * DD + t] + W1[(RR + us) * DD + t] + W1[(RR + TT + ud) * DD + t] + b1[t];
    v = fmaxf(v, 0.f);
  }
  h1[(size_t)c * DD + t] = (__bf16)v;
}

// ---------------------------------------------------------------------------
// Per (b,i): gamma_j = q_i . (k_dst_j + rk[c_ij]) / sqrt(D)  (masked by adj),
// softmax over j, agg_i = S*m_src_i + sum_j alpha_j * rm[c_ij].
// rm/rk are 1MB tables -> L2-resident gathers. One 256-thread block per row.
// ---------------------------------------------------------------------------
__global__ __launch_bounds__(256)
void k_attn_agg(const float* __restrict__ q, const float* __restrict__ kdst,
                const float* __restrict__ msrc, const float* __restrict__ rm,
                const float* __restrict__ rk, const int* __restrict__ adj,
                const int* __restrict__ et, const int* __restrict__ uts,
                const int* __restrict__ utd, __bf16* __restrict__ aggbf) {
  __shared__ float qsh[DD];
  __shared__ float gam[NN];
  __shared__ int   cid[NN];
  __shared__ int   msk[NN];

  const int bi = blockIdx.x;          // b*N + i
  const int b = bi / NN, i = bi % NN;
  const int t = threadIdx.x;

  qsh[t] = q[(size_t)bi * DD + t];
  if (t < NN) {
    const size_t pij = (size_t)(b * NN + i) * NN + t;
    msk[t] = adj[pij];
    cid[t] = (et[pij] * TT + uts[b * NN + i]) * TT + utd[b * NN + t];
  }
  __syncthreads();

  const int lane = t & 31, wave = t >> 5;   // wave32: 8 waves, 16 j's each
  for (int jj = 0; jj < 16; ++jj) {
    const int j = wave * 16 + jj;
    const float* kr = kdst + (size_t)(b * NN + j) * DD;
    const float* rr = rk + (size_t)cid[j] * DD;
    float s = 0.f;
#pragma unroll
    for (int p = 0; p < DD / 32; ++p) {
      const int d = lane + p * 32;
      s += qsh[d] * (kr[d] + rr[d]);
    }
#pragma unroll
    for (int off = 16; off > 0; off >>= 1) s += __shfl_xor(s, off, 32);
    if (lane == 0) gam[j] = msk[j] ? s * 0.0625f : -1e30f;  // 1/sqrt(256)
  }
  __syncthreads();

  float gmax = -1e30f;
  for (int j = 0; j < NN; ++j) gmax = fmaxf(gmax, gam[j]);
  const float S = (gmax <= -1e29f) ? 0.f : 1.f;  // row with no edges -> alpha = 0
  float e = 0.f;
  if (t < NN) e = (S > 0.f) ? __expf(gam[t] - gmax) : 0.f;
  __syncthreads();
  if (t < NN) gam[t] = e;
  __syncthreads();
  float ssum = 0.f;
  for (int j = 0; j < NN; ++j) ssum += gam[j];
  const float inv = (ssum > 0.f) ? 1.f / ssum : 0.f;
  __syncthreads();
  if (t < NN) gam[t] = e * inv;
  __syncthreads();

  float acc = 0.f;
  for (int j = 0; j < NN; ++j)
    acc += gam[j] * rm[(size_t)cid[j] * DD + t];   // coalesced 1KB row gathers
  const float outv = S * msrc[(size_t)bi * DD + t] + acc;
  aggbf[(size_t)bi * DD + t] = (__bf16)outv;
}

// BatchNorm training statistics over MN=1024 rows, one block per column.
__global__ __launch_bounds__(256)
void k_bn_stats(const float* __restrict__ t1, float* __restrict__ mu,
                float* __restrict__ rv) {
  __shared__ float s1[256], s2[256];
  const int c = blockIdx.x, t = threadIdx.x;
  float a1 = 0.f, a2 = 0.f;
  for (int r = t; r < MN; r += 256) {
    const float v = t1[(size_t)r * DD + c];
    a1 += v; a2 += v * v;
  }
  s1[t] = a1; s2[t] = a2;
  __syncthreads();
  for (int off = 128; off > 0; off >>= 1) {
    if (t < off) { s1[t] += s1[t + off]; s2[t] += s2[t + off]; }
    __syncthreads();
  }
  if (t == 0) {
    const float m   = s1[0] / MN;
    const float var = s2[0] / MN - m * m;   // biased, like torch BN
    mu[c] = m;
    rv[c] = rsqrtf(var + 1e-5f);
  }
}

__global__ void k_bn_apply(const float* __restrict__ t1, const float* __restrict__ mu,
                           const float* __restrict__ rv, const float* __restrict__ g,
                           const float* __restrict__ bta, __bf16* __restrict__ out) {
  const int r = blockIdx.x, c = threadIdx.x;
  float v = t1[(size_t)r * DD + c];
  v = (v - mu[c]) * rv[c] * g[c] + bta[c];
  out[(size_t)r * DD + c] = (__bf16)fmaxf(v, 0.f);
}

// ---------------------------------------------------------------------------
extern "C" void kernel_launch(void* const* d_in, const int* in_sizes, int n_in,
                              void* d_out, int out_size, void* d_ws, size_t ws_size,
                              hipStream_t stream) {
  (void)in_sizes; (void)n_in; (void)out_size; (void)ws_size;

  const float* x    = (const float*)d_in[0];
  const int*   adj  = (const int*)d_in[1];
  const int*   et   = (const int*)d_in[2];
  const int*   uts  = (const int*)d_in[3];
  const int*   utd  = (const int*)d_in[4];
  const float* frW1 = (const float*)d_in[5];
  const float* frb1 = (const float*)d_in[6];
  const float* frW2 = (const float*)d_in[7];
  const float* frb2 = (const float*)d_in[8];
  const float* fmW  = (const float*)d_in[9];
  const float* fmb  = (const float*)d_in[10];
  const float* fqW  = (const float*)d_in[11];
  const float* fqb  = (const float*)d_in[12];
  const float* fkW  = (const float*)d_in[13];
  const float* fkb  = (const float*)d_in[14];
  const float* fhW1 = (const float*)d_in[15];
  const float* fhb1 = (const float*)d_in[16];
  const float* bng  = (const float*)d_in[17];
  const float* bnb  = (const float*)d_in[18];
  const float* fhW2 = (const float*)d_in[19];
  const float* fhb2 = (const float*)d_in[20];

  // workspace carve-out (~10 MB total)
  size_t off = 0;
  auto alloc = [&](size_t bytes) -> void* {
    void* p = (char*)d_ws + off;
    off += (bytes + 255) & ~(size_t)255;
    return p;
  };
  __bf16* h1bf  = (__bf16*)alloc((size_t)NCP * DD * 2);
  __bf16* rbf   = (__bf16*)alloc((size_t)NCP * DD * 2);
  float*  rm    = (float*) alloc((size_t)NCP * DD * 4);
  float*  rk    = (float*) alloc((size_t)NCP * DD * 4);
  __bf16* frW2t = (__bf16*)alloc((size_t)DD * DD * 2);
  __bf16* fmW1t = (__bf16*)alloc((size_t)DD * DD * 2);
  __bf16* fmW2t = (__bf16*)alloc((size_t)DD * DD * 2);
  __bf16* fqWt  = (__bf16*)alloc((size_t)DD * DD * 2);
  __bf16* fkW1t = (__bf16*)alloc((size_t)DD * DD * 2);
  __bf16* fkW2t = (__bf16*)alloc((size_t)DD * DD * 2);
  __bf16* fhW1t = (__bf16*)alloc((size_t)DD * DD * 2);
  __bf16* fhW2t = (__bf16*)alloc((size_t)DD * DD * 2);
  __bf16* xbf   = (__bf16*)alloc((size_t)MN * DD * 2);
  float*  qv    = (float*) alloc((size_t)MN * DD * 4);
  float*  msrc  = (float*) alloc((size_t)MN * DD * 4);
  float*  kdst  = (float*) alloc((size_t)MN * DD * 4);
  __bf16* aggbf = (__bf16*)alloc((size_t)MN * DD * 2);
  float*  t1    = (float*) alloc((size_t)MN * DD * 4);
  __bf16* t1bf  = (__bf16*)alloc((size_t)MN * DD * 2);
  float*  mu    = (float*) alloc((size_t)DD * 4);
  float*  rv    = (float*) alloc((size_t)DD * 4);

  const dim3 gblk(32, 8);

  // weight prep: transpose to [out][in] bf16 so the WMMA B-operand packs
  // two consecutive K values per dword
  k_transpose_bf16<<<DD, DD, 0, stream>>>(frW2, 0,   frW2t);
  k_transpose_bf16<<<DD, DD, 0, stream>>>(fmW,  0,   fmW1t);
  k_transpose_bf16<<<DD, DD, 0, stream>>>(fmW,  DD,  fmW2t);
  k_transpose_bf16<<<DD, DD, 0, stream>>>(fqW,  0,   fqWt);
  k_transpose_bf16<<<DD, DD, 0, stream>>>(fkW,  0,   fkW1t);
  k_transpose_bf16<<<DD, DD, 0, stream>>>(fkW,  DD,  fkW2t);
  k_transpose_bf16<<<DD, DD, 0, stream>>>(fhW1, 0,   fhW1t);
  k_transpose_bf16<<<DD, DD, 0, stream>>>(fhW2, 0,   fhW2t);
  k_cvt_bf16<<<(MN * DD + 255) / 256, 256, 0, stream>>>(x, xbf, MN * DD);

  // combo tables: 1000 distinct (edge,src,dst) triples instead of 131072 pairs
  k_build_h1<<<NCP, DD, 0, stream>>>(frW1, frb1, h1bf);
  k_gemm_bf16<<<dim3(NCP / 16, 2), gblk, 0, stream>>>(h1bf, frW2t, frb2, nullptr, nullptr, rbf);
  k_gemm_bf16<<<dim3(NCP / 16, 2), gblk, 0, stream>>>(rbf,  fmW2t, nullptr, nullptr, rm, nullptr);
  k_gemm_bf16<<<dim3(NCP / 16, 2), gblk, 0, stream>>>(rbf,  fkW2t, nullptr, nullptr, rk, nullptr);

  // per-node projections (biases folded so attn pass only adds table rows)
  k_gemm_bf16<<<dim3(MN / 16, 2), gblk, 0, stream>>>(xbf, fqWt,  fqb, nullptr, qv,   nullptr);
  k_gemm_bf16<<<dim3(MN / 16, 2), gblk, 0, stream>>>(xbf, fmW1t, fmb, nullptr, msrc, nullptr);
  k_gemm_bf16<<<dim3(MN / 16, 2), gblk, 0, stream>>>(xbf, fkW1t, fkb, nullptr, kdst, nullptr);

  // attention + aggregation against L2-resident combo tables
  k_attn_agg<<<MN, DD, 0, stream>>>(qv, kdst, msrc, rm, rk, adj, et, uts, utd, aggbf);

  // f_h: Linear -> BN(train stats) -> ReLU -> Linear, + residual into d_out
  k_gemm_bf16<<<dim3(MN / 16, 2), gblk, 0, stream>>>(aggbf, fhW1t, fhb1, nullptr, t1, nullptr);
  k_bn_stats<<<DD, 256, 0, stream>>>(t1, mu, rv);
  k_bn_apply<<<MN, DD, 0, stream>>>(t1, mu, rv, bng, bnb, t1bf);
  k_gemm_bf16<<<dim3(MN / 16, 2), gblk, 0, stream>>>(t1bf, fhW2t, fhb2, x, (float*)d_out, nullptr);
}
